// HhyperLearningLayer_45380624449661
// MI455X (gfx1250) — matrix operand order repaired and checked by reading, not compiled, not run on hardware
//
#include <hip/hip_runtime.h>
#include <hip/hip_bf16.h>

typedef __bf16 bf16_t;
typedef __attribute__((ext_vector_type(16))) __bf16 v16bf;
typedef __attribute__((ext_vector_type(8)))  __bf16 v8bf;
typedef __attribute__((ext_vector_type(4)))  __bf16 v4bf;
typedef __attribute__((ext_vector_type(8)))  float  v8f;

#define DIM_   1024
#define NSEQ   1024
#define BATCH  8
#define HEADS  16
#define DHEAD  64
static constexpr float ATT_SCALE = 0.125f;  // 64^-0.5

// ---------------------------------------------------------------------------
// CDNA5 primitives
// ---------------------------------------------------------------------------
__device__ __forceinline__ v8f wmma_bf16(v16bf a, v16bf b, v8f c) {
  return __builtin_amdgcn_wmma_f32_16x16x32_bf16(
      false, a, false, b, (short)0, c, false, false);
}

// 16B global -> LDS async copy (ASYNCcnt-tracked). LDS offset = low 32 bits of
// the flat shared address (LDS aperture base occupies addr[63:32]).
__device__ __forceinline__ void async_cp16(void* lds_ptr, const void* gptr) {
  unsigned int       lo = (unsigned int)(unsigned long long)(uintptr_t)lds_ptr;
  unsigned long long ga = (unsigned long long)(uintptr_t)gptr;
  asm volatile("global_load_async_to_lds_b128 %0, %1, off"
               :: "v"(lo), "v"(ga) : "memory");
}
__device__ __forceinline__ void wait_async0() {
  asm volatile("s_wait_asynccnt 0" ::: "memory");
}

__device__ __forceinline__ v8f vzero8() {
  v8f z = {0.f, 0.f, 0.f, 0.f, 0.f, 0.f, 0.f, 0.f};
  return z;
}

// A-fragment (16xK bf16, row-major [row][k]): lane<16 -> K {kh*8..+7, 16+kh*8..+7}
__device__ __forceinline__ v16bf frag_a(const bf16_t* base, int stride, int lane) {
  const int r  = lane & 15;
  const int kh = (lane >> 4) & 1;
  const bf16_t* p = base + r * stride + kh * 8;
  v8bf lo = *(const v8bf*)(p);
  v8bf hi = *(const v8bf*)(p + 16);
  return __builtin_shufflevector(lo, hi, 0,1,2,3,4,5,6,7,8,9,10,11,12,13,14,15);
}

// B-fragment (Kx16 bf16, stored transposed [n][k]): lane<16 -> K {kh*8..+7, 16+kh*8..+7}
__device__ __forceinline__ v16bf frag_b(const bf16_t* base, int stride, int lane) {
  const int r  = lane & 15;
  const int kh = (lane >> 4) & 1;
  const bf16_t* p = base + r * stride + kh * 8;
  v8bf lo = *(const v8bf*)(p);
  v8bf hi = *(const v8bf*)(p + 16);
  return __builtin_shufflevector(lo, hi, 0,1,2,3,4,5,6,7,8,9,10,11,12,13,14,15);
}

// ---------------------------------------------------------------------------
// Kernel 0a: elementwise f32 -> bf16 (activations)
// ---------------------------------------------------------------------------
__global__ __launch_bounds__(256) void cvt_kernel(
    const float* __restrict__ src, bf16_t* __restrict__ dst, int n4) {
  int i = blockIdx.x * 256 + threadIdx.x;
  if (i < n4) {
    const float4 v = ((const float4*)src)[i];
    v4bf pk = {(bf16_t)v.x, (bf16_t)v.y, (bf16_t)v.z, (bf16_t)v.w};
    ((v4bf*)dst)[i] = pk;
  }
}

// ---------------------------------------------------------------------------
// Kernel 0b: weight transpose + convert: Wt[n][k] = (bf16)W[k][n], 1024x1024
// ---------------------------------------------------------------------------
__global__ __launch_bounds__(256) void wtrans_kernel(
    const float* __restrict__ W, bf16_t* __restrict__ Wt) {
  __shared__ bf16_t T[64 * 72];          // [k][n] tile, padded
  const int k0 = blockIdx.x * 64, n0 = blockIdx.y * 64;
  const int tid = threadIdx.x;
#pragma unroll
  for (int i = 0; i < 4; ++i) {
    int linear = (i * 256 + tid) * 4;    // 64x64 elements
    int r = linear >> 6, c = linear & 63;
    const float4 v = *(const float4*)(W + (size_t)(k0 + r) * DIM_ + n0 + c);
    v4bf pk = {(bf16_t)v.x, (bf16_t)v.y, (bf16_t)v.z, (bf16_t)v.w};
    *(v4bf*)&T[r * 72 + c] = pk;
  }
  __syncthreads();
#pragma unroll
  for (int i = 0; i < 2; ++i) {
    int linear = (i * 256 + tid) * 8;
    int r = linear >> 6, c = linear & 63;   // r: n index, c: k start
    v8bf pk;
#pragma unroll
    for (int e = 0; e < 8; ++e) pk[e] = T[(c + e) * 72 + r];
    *(v8bf*)(Wt + (size_t)(n0 + r) * DIM_ + k0 + c) = pk;
  }
}

// ---------------------------------------------------------------------------
// Kernel 1: fused 5-way projection GEMM, all-bf16 async staging.
// Block 128x128, 8 waves (4x2), wave 32x64, K-step 64 (16 WMMA / barrier).
// z==2,4 (V projections) store transposed [b,h,d,n] for flash V staging.
// ---------------------------------------------------------------------------
__global__ __launch_bounds__(256) void proj_gemm_kernel(
    const bf16_t* __restrict__ htb, const bf16_t* __restrict__ hab,
    const bf16_t* __restrict__ hvb,
    const bf16_t* __restrict__ WqT,  const bf16_t* __restrict__ WktaT,
    const bf16_t* __restrict__ WvtaT, const bf16_t* __restrict__ WktvT,
    const bf16_t* __restrict__ WvtvT,
    bf16_t* __restrict__ Qw,  bf16_t* __restrict__ Kta, bf16_t* __restrict__ Vta,
    bf16_t* __restrict__ Ktv, bf16_t* __restrict__ Vtv)
{
  __shared__ bf16_t As[128 * 72];   // [m][k], K-step 64, pad stride 72
  __shared__ bf16_t Bs[128 * 72];   // [n][k]

  const int z = blockIdx.z;
  const bf16_t* A  = (z == 0) ? htb : (z <= 2 ? hab : hvb);
  const bf16_t* Wt = (z == 0) ? WqT : (z == 1 ? WktaT : (z == 2 ? WvtaT
                                     : (z == 3 ? WktvT : WvtvT)));
  bf16_t* D = (z == 0) ? Qw : (z == 1 ? Kta : (z == 2 ? Vta
                              : (z == 3 ? Ktv : Vtv)));
  const bool vtrans = (z == 2) || (z == 4);

  const int tid  = threadIdx.x;
  const int lane = tid & 31;
  const int w    = tid >> 5;
  const int wm   = w >> 1, wn = w & 1;
  const int m0   = blockIdx.x * 128;
  const int n0   = blockIdx.y * 128;

  v8f acc[2][4];
#pragma unroll
  for (int i = 0; i < 2; ++i)
#pragma unroll
    for (int j = 0; j < 4; ++j) acc[i][j] = vzero8();

  for (int kt = 0; kt < DIM_ / 64; ++kt) {
    const int k0 = kt * 64;
    __syncthreads();
    // Async-stage 128x64 bf16 tiles of A and W^T (8x 16B per thread)
#pragma unroll
    for (int i = 0; i < 4; ++i) {
      int linear = i * 256 + tid;          // 0..1023
      int row = linear >> 3, seg = (linear & 7) * 8;
      async_cp16(&As[row * 72 + seg], A  + (size_t)(m0 + row) * DIM_ + k0 + seg);
      async_cp16(&Bs[row * 72 + seg], Wt + (size_t)(n0 + row) * DIM_ + k0 + seg);
    }
    wait_async0();
    __syncthreads();

#pragma unroll
    for (int ks = 0; ks < 2; ++ks) {
      v16bf af[2], bfr[4];
#pragma unroll
      for (int it = 0; it < 2; ++it)
        af[it] = frag_a(&As[(wm * 32 + it * 16) * 72 + ks * 32], 72, lane);
#pragma unroll
      for (int jt = 0; jt < 4; ++jt)
        bfr[jt] = frag_b(&Bs[(wn * 64 + jt * 16) * 72 + ks * 32], 72, lane);
#pragma unroll
      for (int it = 0; it < 2; ++it)
#pragma unroll
        for (int jt = 0; jt < 4; ++jt)
          acc[it][jt] = wmma_bf16(af[it], bfr[jt], acc[it][jt]);
    }
  }

  const int rsel = (lane & 16) >> 1;
#pragma unroll
  for (int it = 0; it < 2; ++it)
#pragma unroll
    for (int jt = 0; jt < 4; ++jt)
#pragma unroll
      for (int j = 0; j < 8; ++j) {
        int gm = m0 + wm * 32 + it * 16 + j + rsel;
        int gn = n0 + wn * 64 + jt * 16 + (lane & 15);
        int b  = gm >> 10, ns = gm & 1023;
        int hh = gn >> 6,  dd = gn & 63;
        size_t off = vtrans
            ? ((((size_t)b * HEADS + hh) * DHEAD + dd) * NSEQ + ns)   // [b,h,d,n]
            : ((((size_t)b * HEADS + hh) * NSEQ + ns) * DHEAD + dd);  // [b,h,n,d]
        D[off] = (bf16_t)acc[it][jt][j];
      }
}

// ---------------------------------------------------------------------------
// Kernel 2: dual-branch flash attention; S^T = K*Q^T trick keeps softmax
// in-lane; K and V^T chunks staged with async copies.
// ---------------------------------------------------------------------------
__global__ __launch_bounds__(256) void flash_attn_kernel(
    const bf16_t* __restrict__ Qw,
    const bf16_t* __restrict__ Kta, const bf16_t* __restrict__ VtaT,
    const bf16_t* __restrict__ Ktv, const bf16_t* __restrict__ VtvT,
    bf16_t* __restrict__ Mg)
{
  __shared__ bf16_t Ks[32 * 72];    // [key][d]
  __shared__ bf16_t Vs[64 * 40];    // [d][key]

  const int bh   = blockIdx.x;
  const int b    = bh >> 4, h = bh & 15;
  const int tid  = threadIdx.x;
  const int lane = tid & 31;
  const int w    = tid >> 5;
  const int q0w  = blockIdx.y * 128 + w * 16;
  const int rsel = (lane & 16) >> 1;

  const bf16_t* qbase = Qw + ((size_t)bh * NSEQ + q0w) * DHEAD;
  v16bf qf[2];
#pragma unroll
  for (int ds_ = 0; ds_ < 2; ++ds_)
    qf[ds_] = frag_b(qbase + ds_ * 32, DHEAD, lane);

  v8f osum[4];
#pragma unroll
  for (int nt = 0; nt < 4; ++nt) osum[nt] = vzero8();

  for (int br = 0; br < 2; ++br) {
    const bf16_t* Kp = br ? Ktv  : Kta;
    const bf16_t* Vt = br ? VtvT : VtaT;
    float mrun = -1e30f, lrun = 0.f;
    v8f o[4];
#pragma unroll
    for (int nt = 0; nt < 4; ++nt) o[nt] = vzero8();

    for (int ch = 0; ch < NSEQ / 32; ++ch) {
      __syncthreads();
      {
        int key = tid >> 3, dblk = (tid & 7) * 8;   // K: 32x64
        async_cp16(&Ks[key * 72 + dblk],
                   Kp + ((size_t)bh * NSEQ + ch * 32 + key) * DHEAD + dblk);
        int dr = tid >> 2, seg = (tid & 3) * 8;     // V^T: 64 rows x 32 keys
        async_cp16(&Vs[dr * 40 + seg],
                   Vt + ((size_t)bh * DHEAD + dr) * NSEQ + ch * 32 + seg);
      }
      wait_async0();
      __syncthreads();

      v8f s0 = vzero8(), s1 = vzero8();
#pragma unroll
      for (int ds_ = 0; ds_ < 2; ++ds_) {
        v16bf a0 = frag_a(&Ks[ds_ * 32],           72, lane);
        v16bf a1 = frag_a(&Ks[16 * 72 + ds_ * 32], 72, lane);
        s0 = wmma_bf16(a0, qf[ds_], s0);
        s1 = wmma_bf16(a1, qf[ds_], s1);
      }

      float mloc = -1e30f;
#pragma unroll
      for (int j = 0; j < 8; ++j) mloc = fmaxf(mloc, fmaxf(s0[j], s1[j]));
      mloc *= ATT_SCALE;
      mloc = fmaxf(mloc, __shfl_xor(mloc, 16, 32));
      float mnew = fmaxf(mrun, mloc);
      float corr = __expf(mrun - mnew);

      v16bf pf;
      float psum = 0.f;
#pragma unroll
      for (int j = 0; j < 8; ++j) {
        float p0 = __expf(s0[j] * ATT_SCALE - mnew);
        float p1 = __expf(s1[j] * ATT_SCALE - mnew);
        psum += p0 + p1;
        pf[j]     = (bf16_t)p0;
        pf[j + 8] = (bf16_t)p1;
      }
      psum += __shfl_xor(psum, 16, 32);
      lrun = lrun * corr + psum;
      mrun = mnew;

      float cj[8];
#pragma unroll
      for (int j = 0; j < 8; ++j) cj[j] = __shfl(corr, j + rsel, 32);
#pragma unroll
      for (int nt = 0; nt < 4; ++nt)
#pragma unroll
        for (int j = 0; j < 8; ++j) o[nt][j] *= cj[j];

#pragma unroll
      for (int nt = 0; nt < 4; ++nt) {
        v16bf vb = frag_b(&Vs[(nt * 16) * 40], 40, lane);
        o[nt] = wmma_bf16(pf, vb, o[nt]);
      }
    }

    float inv = 1.f / lrun;
    float ij_[8];
#pragma unroll
    for (int j = 0; j < 8; ++j) ij_[j] = __shfl(inv, j + rsel, 32);
#pragma unroll
    for (int nt = 0; nt < 4; ++nt)
#pragma unroll
      for (int j = 0; j < 8; ++j) osum[nt][j] += o[nt][j] * ij_[j];
  }

#pragma unroll
  for (int nt = 0; nt < 4; ++nt) {
    int dd = nt * 16 + (lane & 15);
#pragma unroll
    for (int j = 0; j < 8; ++j) {
      int q = q0w + j + rsel;
      Mg[(((size_t)b * NSEQ + q) * HEADS + h) * DHEAD + dd] = (bf16_t)osum[nt][j];
    }
  }
}

// ---------------------------------------------------------------------------
// Kernel 3: output GEMM + residual + bias, f32 out; async bf16 staging.
// ---------------------------------------------------------------------------
__global__ __launch_bounds__(256) void out_gemm_kernel(
    const bf16_t* __restrict__ Mg, const bf16_t* __restrict__ WoutT,
    const float* __restrict__ h_hyper, const float* __restrict__ bout,
    float* __restrict__ out)
{
  __shared__ bf16_t As[128 * 72];
  __shared__ bf16_t Bs[128 * 72];

  const int tid  = threadIdx.x;
  const int lane = tid & 31;
  const int w    = tid >> 5;
  const int wm   = w >> 1, wn = w & 1;
  const int m0   = blockIdx.x * 128;
  const int n0   = blockIdx.y * 128;

  v8f acc[2][4];
#pragma unroll
  for (int i = 0; i < 2; ++i)
#pragma unroll
    for (int j = 0; j < 4; ++j) acc[i][j] = vzero8();

  for (int kt = 0; kt < DIM_ / 64; ++kt) {
    const int k0 = kt * 64;
    __syncthreads();
#pragma unroll
    for (int i = 0; i < 4; ++i) {
      int linear = i * 256 + tid;
      int row = linear >> 3, seg = (linear & 7) * 8;
      async_cp16(&As[row * 72 + seg], Mg    + (size_t)(m0 + row) * DIM_ + k0 + seg);
      async_cp16(&Bs[row * 72 + seg], WoutT + (size_t)(n0 + row) * DIM_ + k0 + seg);
    }
    wait_async0();
    __syncthreads();

#pragma unroll
    for (int ks = 0; ks < 2; ++ks) {
      v16bf af[2], bfr[4];
#pragma unroll
      for (int it = 0; it < 2; ++it)
        af[it] = frag_a(&As[(wm * 32 + it * 16) * 72 + ks * 32], 72, lane);
#pragma unroll
      for (int jt = 0; jt < 4; ++jt)
        bfr[jt] = frag_b(&Bs[(wn * 64 + jt * 16) * 72 + ks * 32], 72, lane);
#pragma unroll
      for (int it = 0; it < 2; ++it)
#pragma unroll
        for (int jt = 0; jt < 4; ++jt)
          acc[it][jt] = wmma_bf16(af[it], bfr[jt], acc[it][jt]);
    }
  }

  const int rsel = (lane & 16) >> 1;
#pragma unroll
  for (int it = 0; it < 2; ++it)
#pragma unroll
    for (int jt = 0; jt < 4; ++jt) {
      int gn = n0 + wn * 64 + jt * 16 + (lane & 15);
      float bb = bout[gn];
#pragma unroll
      for (int j = 0; j < 8; ++j) {
        int gm = m0 + wm * 32 + it * 16 + j + rsel;
        size_t off = (size_t)gm * DIM_ + gn;
        out[off] = acc[it][jt][j] + h_hyper[off] + bb;
      }
    }
}

// ---------------------------------------------------------------------------
extern "C" void kernel_launch(void* const* d_in, const int* in_sizes, int n_in,
                              void* d_out, int out_size, void* d_ws, size_t ws_size,
                              hipStream_t stream) {
  const float* h_t     = (const float*)d_in[0];
  const float* h_a     = (const float*)d_in[1];
  const float* h_v     = (const float*)d_in[2];
  const float* h_hyper = (const float*)d_in[3];
  const float* Wq      = (const float*)d_in[4];
  const float* Wk_ta   = (const float*)d_in[5];
  const float* Wk_tv   = (const float*)d_in[6];
  const float* Wv_ta   = (const float*)d_in[7];
  const float* Wv_tv   = (const float*)d_in[8];
  const float* Wout    = (const float*)d_in[9];
  const float* bout    = (const float*)d_in[10];

  char* ws = (char*)d_ws;
  const size_t ACT  = (size_t)BATCH * NSEQ * DIM_ * sizeof(bf16_t);          // 16 MB
  const size_t WMAT = (size_t)DIM_ * DIM_ * sizeof(bf16_t);                  //  2 MB
  const size_t PROJ = (size_t)BATCH * HEADS * NSEQ * DHEAD * sizeof(bf16_t); // 16 MB

  bf16_t* htb   = (bf16_t*)(ws);
  bf16_t* hab   = (bf16_t*)(ws + ACT);
  bf16_t* hvb   = (bf16_t*)(ws + 2 * ACT);
  char*   wbase = ws + 3 * ACT;
  bf16_t* WqT   = (bf16_t*)(wbase + 0 * WMAT);
  bf16_t* WktaT = (bf16_t*)(wbase + 1 * WMAT);
  bf16_t* WvtaT = (bf16_t*)(wbase + 2 * WMAT);
  bf16_t* WktvT = (bf16_t*)(wbase + 3 * WMAT);
  bf16_t* WvtvT = (bf16_t*)(wbase + 4 * WMAT);
  bf16_t* WoutT = (bf16_t*)(wbase + 5 * WMAT);
  char*   pbase = wbase + 6 * WMAT;
  bf16_t* Qw    = (bf16_t*)(pbase + 0 * PROJ);
  bf16_t* Kta   = (bf16_t*)(pbase + 1 * PROJ);
  bf16_t* VtaT  = (bf16_t*)(pbase + 2 * PROJ);
  bf16_t* Ktv   = (bf16_t*)(pbase + 3 * PROJ);
  bf16_t* VtvT  = (bf16_t*)(pbase + 4 * PROJ);
  bf16_t* Mg    = (bf16_t*)(pbase + 5 * PROJ);

  dim3 blk(256);
  const int n4 = (BATCH * NSEQ * DIM_) / 4;   // 2,097,152
  cvt_kernel<<<dim3(n4 / 256), blk, 0, stream>>>(h_t, htb, n4);
  cvt_kernel<<<dim3(n4 / 256), blk, 0, stream>>>(h_a, hab, n4);
  cvt_kernel<<<dim3(n4 / 256), blk, 0, stream>>>(h_v, hvb, n4);
  wtrans_kernel<<<dim3(16, 16), blk, 0, stream>>>(Wq,    WqT);
  wtrans_kernel<<<dim3(16, 16), blk, 0, stream>>>(Wk_ta, WktaT);
  wtrans_kernel<<<dim3(16, 16), blk, 0, stream>>>(Wv_ta, WvtaT);
  wtrans_kernel<<<dim3(16, 16), blk, 0, stream>>>(Wk_tv, WktvT);
  wtrans_kernel<<<dim3(16, 16), blk, 0, stream>>>(Wv_tv, WvtvT);
  wtrans_kernel<<<dim3(16, 16), blk, 0, stream>>>(Wout,  WoutT);

  proj_gemm_kernel<<<dim3(64, 8, 5), blk, 0, stream>>>(
      htb, hab, hvb, WqT, WktaT, WvtaT, WktvT, WvtvT, Qw, Kta, VtaT, Ktv, VtvT);
  flash_attn_kernel<<<dim3(BATCH * HEADS, NSEQ / 128), blk, 0, stream>>>(
      Qw, Kta, VtaT, Ktv, VtvT, Mg);
  out_gemm_kernel<<<dim3(64, 8), blk, 0, stream>>>(
      Mg, WoutT, h_hyper, bout, (float*)d_out);
}
